// LegacyCTNNJastrow_9311489098278
// MI455X (gfx1250) — matrix operand order repaired and checked by reading, not compiled, not run on hardware
//
#include <hip/hip_runtime.h>
#include <math.h>

typedef _Float16 h16;
typedef __attribute__((ext_vector_type(16))) _Float16 v16h;
typedef __attribute__((ext_vector_type(8)))  float    v8f;

#define B_SZ   1024
#define NPART  16
#define NEDGE  240
#define HID    128
#define KSPLIT 8

__device__ __forceinline__ float gelu_f(float v) {
    // exact GELU: 0.5*x*(1+erf(x/sqrt(2)))
    return 0.5f * v * (1.0f + erff(v * 0.70710678118654752f));
}

// ---- WMMA fragment helpers (16x16x32 f16, wave32) -------------------------
// A-matrix 16x32 f16 layout (ISA 7.12.2): lane m = lane&15, half = lane>>4;
// elements 0..7 -> k = k0 + half*8 + i ; elements 8..15 -> k = k0 + 16 + half*8 + i
__device__ __forceinline__ v16h fragA(const h16* row, int k0, int half) {
    v16h f;
    int b1 = k0 + half * 8;
    int b2 = k0 + 16 + half * 8;
#pragma unroll
    for (int i = 0; i < 8; ++i) f[i] = row[b1 + i];
#pragma unroll
    for (int i = 0; i < 8; ++i) f[8 + i] = row[b2 + i];
    return f;
}

// B-matrix 32x16 f16: lane holds column n = lane&15; elements 0..7 -> k = k0 + half*8 + i;
// elements 8..15 -> k = k0 + 16 + half*8 + i (same k order as fragA).
// B[k][n] == W[n][k] (weights are row-major (out,in)), so we read a row of W.
__device__ __forceinline__ v16h fragB(const h16* wrow, int k0, int half) {
    v16h f;
    int b1 = k0 + half * 8;
    int b2 = k0 + 16 + half * 8;
#pragma unroll
    for (int i = 0; i < 8; ++i) f[i] = wrow[b1 + i];
#pragma unroll
    for (int i = 0; i < 8; ++i) f[8 + i] = wrow[b2 + i];
    return f;
}

__device__ __forceinline__ v8f wmma_f16(v16h a, v16h b, v8f c) {
    return __builtin_amdgcn_wmma_f32_16x16x32_f16(false, a, false, b,
                                                  (short)0, c, false, false);
}

// ---- K0: f32 -> f16 weight conversion (strided rows) ----------------------
__global__ void k_cvt2d(const float* __restrict__ src, h16* __restrict__ dst,
                        int rows, int cols, int src_stride) {
    int total = rows * cols;
    for (int i = blockIdx.x * blockDim.x + threadIdx.x; i < total;
         i += gridDim.x * blockDim.x) {
        int r = i / cols, c = i - r * cols;
        dst[i] = (h16)src[(size_t)r * src_stride + c];
    }
}

__global__ void k_zero(float* __restrict__ p, int n) {
    int i = blockIdx.x * blockDim.x + threadIdx.x;
    if (i < n) p[i] = 0.0f;
}

// ---- K1: node embedding  h_v = W_node @ [x,spin] + b ----------------------
__global__ void k_init_nodes(const float* __restrict__ x,
                             const int* __restrict__ spin,
                             const float* __restrict__ Wn,
                             const float* __restrict__ bn,
                             h16* __restrict__ h_v) {
    int b = blockIdx.x;      // 1024
    int n = threadIdx.x;     // 128
    const float* w = Wn + n * 4;
#pragma unroll 4
    for (int p = 0; p < NPART; ++p) {
        const float* xp = x + ((size_t)b * NPART + p) * 3;
        float sp = (float)spin[b * NPART + p];
        float y = bn[n] + w[0] * xp[0] + w[1] * xp[1] + w[2] * xp[2] + w[3] * sp;
        h_v[((size_t)b * NPART + p) * HID + n] = (h16)y;
    }
}

// ---- K2: edge init  h_e = lin2(gelu(lin1(feat))) --------------------------
__global__ void k_init_edges(const float* __restrict__ x,
                             const float* __restrict__ We1,
                             const float* __restrict__ be1,
                             const h16* __restrict__ We2f,
                             const float* __restrict__ be2,
                             h16* __restrict__ h_e) {
    __shared__ h16 tbuf[16 * HID];
    int b = blockIdx.y, mt = blockIdx.x;       // 15 edge tiles of 16
    for (int idx = threadIdx.x; idx < 16 * HID; idx += blockDim.x) {
        int el = idx >> 7, n = idx & 127;
        int e = mt * 16 + el;
        int s = e / 15, j = e - s * 15;
        int d = j + (j >= s);
        const float* xs = x + ((size_t)b * NPART + s) * 3;
        const float* xd = x + ((size_t)b * NPART + d) * 3;
        float dr0 = xd[0] - xs[0], dr1 = xd[1] - xs[1], dr2 = xd[2] - xs[2];
        float r2 = dr0 * dr0 + dr1 * dr1 + dr2 * dr2;
        float rr = sqrtf(r2 + 1e-12f);
        const float* w = We1 + n * 5;
        float y = be1[n] + w[0]*dr0 + w[1]*dr1 + w[2]*dr2 + w[3]*rr + w[4]*r2;
        tbuf[idx] = (h16)gelu_f(y);
    }
    __syncthreads();
    int wave = threadIdx.x >> 5, lane = threadIdx.x & 31;
    int m = lane & 15, half = lane >> 4;
    int nrow = wave * 16 + (lane & 15);
    v8f c = {};
#pragma unroll
    for (int kc = 0; kc < 4; ++kc)
        c = wmma_f16(fragA(&tbuf[m * HID], kc * 32, half),
                     fragB(&We2f[nrow * HID], kc * 32, half), c);
    h16* out = h_e + (size_t)b * NEDGE * HID + (size_t)mt * 16 * HID;
#pragma unroll
    for (int r = 0; r < 8; ++r) {
        int mm = r + 8 * half;
        out[mm * HID + nrow] = (h16)(c[r] + be2[nrow]);
    }
}

// ---- K3: v2e = h_v @ W_v2e^T ----------------------------------------------
__global__ void k_v2e(const h16* __restrict__ h_v,
                      const h16* __restrict__ Wf,
                      h16* __restrict__ v2e) {
    __shared__ h16 A[16 * HID];
    int b = blockIdx.x;
    for (int idx = threadIdx.x; idx < 16 * HID; idx += blockDim.x)
        A[idx] = h_v[(size_t)b * 2048 + idx];
    __syncthreads();
    int wave = threadIdx.x >> 5, lane = threadIdx.x & 31;
    int m = lane & 15, half = lane >> 4;
    int nrow = wave * 16 + (lane & 15);
    v8f c = {};
#pragma unroll
    for (int kc = 0; kc < 4; ++kc)
        c = wmma_f16(fragA(&A[m * HID], kc * 32, half),
                     fragB(&Wf[nrow * HID], kc * 32, half), c);
#pragma unroll
    for (int r = 0; r < 8; ++r) {
        int mm = r + 8 * half;
        v2e[(size_t)b * 2048 + mm * HID + nrow] = (h16)c[r];
    }
}

// ---- K4: edge update  h_e = MLP([h_e, v2e[src], v2e[dst]]) (in place) -----
__global__ void k_edge_update(const float* __restrict__ b_eu1,
                              const float* __restrict__ b_eu2,
                              const h16* __restrict__ Weu1f,   // (128,384) f16
                              const h16* __restrict__ Weu2f,   // (128,128) f16
                              h16* __restrict__ h_e,
                              const h16* __restrict__ v2e) {
    __shared__ h16 Ahe[16 * HID], Asrc[16 * HID], Adst[16 * HID], Hid[16 * HID];
    int b = blockIdx.y, mt = blockIdx.x;
    h16* he_base = h_e + (size_t)b * NEDGE * HID + (size_t)mt * 16 * HID;
    for (int idx = threadIdx.x; idx < 16 * HID; idx += blockDim.x) {
        int el = idx >> 7, n = idx & 127;
        int e = mt * 16 + el;
        int s = e / 15, j = e - s * 15;
        int d = j + (j >= s);
        Ahe[idx]  = he_base[idx];
        Asrc[idx] = v2e[(size_t)b * 2048 + s * HID + n];
        Adst[idx] = v2e[(size_t)b * 2048 + d * HID + n];
    }
    __syncthreads();
    int wave = threadIdx.x >> 5, lane = threadIdx.x & 31;
    int m = lane & 15, half = lane >> 4;
    int nrow = wave * 16 + (lane & 15);
    v8f c = {};
#pragma unroll
    for (int kc = 0; kc < 4; ++kc) {
        c = wmma_f16(fragA(&Ahe[m * HID],  kc * 32, half),
                     fragB(&Weu1f[nrow * 384 + 0],   kc * 32, half), c);
        c = wmma_f16(fragA(&Asrc[m * HID], kc * 32, half),
                     fragB(&Weu1f[nrow * 384 + 128], kc * 32, half), c);
        c = wmma_f16(fragA(&Adst[m * HID], kc * 32, half),
                     fragB(&Weu1f[nrow * 384 + 256], kc * 32, half), c);
    }
#pragma unroll
    for (int r = 0; r < 8; ++r) {
        int mm = r + 8 * half;
        Hid[mm * HID + nrow] = (h16)gelu_f(c[r] + b_eu1[nrow]);
    }
    __syncthreads();
    v8f c2 = {};
#pragma unroll
    for (int kc = 0; kc < 4; ++kc)
        c2 = wmma_f16(fragA(&Hid[m * HID], kc * 32, half),
                      fragB(&Weu2f[nrow * HID], kc * 32, half), c2);
#pragma unroll
    for (int r = 0; r < 8; ++r) {
        int mm = r + 8 * half;
        he_base[mm * HID + nrow] = (h16)(c2[r] + b_eu2[nrow]);
    }
}

// ---- K5: msg = h_e @ W_e2v^T, agg[dst]+=msg/15, h_v = MLP([h_v,agg]) ------
__global__ void k_e2v_node(const h16* __restrict__ h_e,
                           const h16* __restrict__ We2vf,
                           const h16* __restrict__ Wnu1f,  // (128,256) f16
                           const float* __restrict__ b_nu1,
                           const h16* __restrict__ Wnu2f,
                           const float* __restrict__ b_nu2,
                           h16* __restrict__ h_v) {
    __shared__ float agg[16 * HID];
    __shared__ h16 Ahv[16 * HID], Aagg[16 * HID], Hid[16 * HID];
    int b = blockIdx.x;
    for (int idx = threadIdx.x; idx < 16 * HID; idx += blockDim.x)
        agg[idx] = 0.0f;
    __syncthreads();
    int wave = threadIdx.x >> 5, lane = threadIdx.x & 31;
    int m = lane & 15, half = lane >> 4;
    // 15 M-tiles x 8 N-tiles = 120 tile jobs over 8 waves
    for (int job = wave; job < 120; job += 8) {
        int mt = job >> 3, nt = job & 7;
        const h16* Arow = h_e + (size_t)b * NEDGE * HID + (mt * 16 + m) * HID;
        int nrow = nt * 16 + (lane & 15);
        v8f c = {};
#pragma unroll
        for (int kc = 0; kc < 4; ++kc)
            c = wmma_f16(fragA(Arow, kc * 32, half),
                         fragB(&We2vf[nrow * HID], kc * 32, half), c);
#pragma unroll
        for (int r = 0; r < 8; ++r) {
            int e = mt * 16 + r + 8 * half;
            int s = e / 15, j = e - s * 15;
            int d = j + (j >= s);
            atomicAdd(&agg[d * HID + nrow], c[r] * (1.0f / 15.0f));  // ds_add_f32
        }
    }
    __syncthreads();
    for (int idx = threadIdx.x; idx < 16 * HID; idx += blockDim.x) {
        Ahv[idx]  = h_v[(size_t)b * 2048 + idx];
        Aagg[idx] = (h16)agg[idx];
    }
    __syncthreads();
    int nrow = wave * 16 + (lane & 15);
    v8f c = {};
#pragma unroll
    for (int kc = 0; kc < 4; ++kc) {
        c = wmma_f16(fragA(&Ahv[m * HID],  kc * 32, half),
                     fragB(&Wnu1f[nrow * 256 + 0],   kc * 32, half), c);
        c = wmma_f16(fragA(&Aagg[m * HID], kc * 32, half),
                     fragB(&Wnu1f[nrow * 256 + 128], kc * 32, half), c);
    }
#pragma unroll
    for (int r = 0; r < 8; ++r) {
        int mm = r + 8 * half;
        Hid[mm * HID + nrow] = (h16)gelu_f(c[r] + b_nu1[nrow]);
    }
    __syncthreads();
    v8f c2 = {};
#pragma unroll
    for (int kc = 0; kc < 4; ++kc)
        c2 = wmma_f16(fragA(&Hid[m * HID], kc * 32, half),
                      fragB(&Wnu2f[nrow * HID], kc * 32, half), c2);
#pragma unroll
    for (int r = 0; r < 8; ++r) {
        int mm = r + 8 * half;
        h_v[(size_t)b * 2048 + mm * HID + nrow] = (h16)(c2[r] + b_nu2[nrow]);
    }
}

// ---- K6a: per-row tail features (r_pair, r2_all) --------------------------
__global__ void k_rfeat(const float* __restrict__ x,
                        float* __restrict__ rp, float* __restrict__ r2a) {
    int b = blockIdx.x * blockDim.x + threadIdx.x;
    if (b >= B_SZ) return;
    const float* xb = x + (size_t)b * 48;
    float s = 0.0f;
#pragma unroll
    for (int i = 0; i < 48; ++i) s += xb[i] * xb[i];
    r2a[b] = s;
    float d0 = xb[0] - xb[3], d1 = xb[1] - xb[4], d2 = xb[2] - xb[5];
    rp[b] = sqrtf(d0 * d0 + d1 * d1 + d2 * d2 + 1e-12f);
}

// ---- K6b: split-K readout GEMM  acc += f_in @ Wf1^T (K=32768) -------------
// grid (64 M-tiles, KSPLIT);  512 workgroups to saturate HBM on MI455X.
__global__ void k_final1_split(const h16* __restrict__ h_v,
                               const h16* __restrict__ h_e,
                               const h16* __restrict__ Wf1f,   // (128,32768) f16
                               float* __restrict__ acc) {      // (1024,128) f32
    __shared__ h16 A[16 * HID];
    int bt = blockIdx.x;             // batch tile (16 rows)
    int ks = blockIdx.y;             // K split
    int tid = threadIdx.x;
    int wave = tid >> 5, lane = tid & 31;
    int m = lane & 15, half = lane >> 4;
    int nrow = wave * 16 + (lane & 15);
    v8f c = {};
    const int CHUNKS = 256 / KSPLIT; // 32 k-chunks of 128
    for (int i = 0; i < CHUNKS; ++i) {
        int kc = ks * CHUNKS + i;
        __syncthreads();
        for (int idx = tid; idx < 16 * HID; idx += 256) {
            int mm = idx >> 7, kk = idx & 127;
            int b = bt * 16 + mm;
            int kg = kc * 128 + kk;
            A[idx] = (kg < 2048) ? h_v[(size_t)b * 2048 + kg]
                                 : h_e[(size_t)b * 30720 + (kg - 2048)];
        }
        __syncthreads();
#pragma unroll
        for (int sub = 0; sub < 4; ++sub)
            c = wmma_f16(fragA(&A[m * HID], sub * 32, half),
                         fragB(&Wf1f[(size_t)nrow * 32768 + kc * 128], sub * 32, half), c);
    }
#pragma unroll
    for (int r = 0; r < 8; ++r) {
        int mm = r + 8 * half;
        int b = bt * 16 + mm;
        atomicAdd(&acc[(size_t)b * HID + nrow], c[r]);   // global_atomic_add_f32
    }
}

// ---- K6c: epilogue  h = gelu(acc + bf1 + tail cols) -----------------------
__global__ void k_final1_epi(const float* __restrict__ acc,
                             const float* __restrict__ Wf1,  // (128,32770) f32
                             const float* __restrict__ bf1,
                             const float* __restrict__ rp,
                             const float* __restrict__ r2a,
                             h16* __restrict__ hbuf) {
    int i = blockIdx.x * blockDim.x + threadIdx.x;   // 0 .. 1024*128-1
    int b = i >> 7, n = i & 127;
    float y = acc[i] + bf1[n]
            + Wf1[(size_t)n * 32770 + 32768] * rp[b]
            + Wf1[(size_t)n * 32770 + 32769] * r2a[b];
    hbuf[i] = (h16)gelu_f(y);
}

// ---- K7: h2 = gelu(h @ Wf2^T + bf2); out = h2 @ Wf3^T + bf3 ---------------
__global__ void k_final2(const h16* __restrict__ hbuf,
                         const h16* __restrict__ Wf2f,
                         const float* __restrict__ bf2,
                         const float* __restrict__ Wf3,
                         const float* __restrict__ bf3,
                         float* __restrict__ out) {
    __shared__ h16 A[16 * HID];
    __shared__ float H2[16 * HID];
    int bt = blockIdx.x;
    for (int idx = threadIdx.x; idx < 16 * HID; idx += blockDim.x)
        A[idx] = hbuf[(size_t)(bt * 16 + (idx >> 7)) * HID + (idx & 127)];
    __syncthreads();
    int wave = threadIdx.x >> 5, lane = threadIdx.x & 31;
    int m = lane & 15, half = lane >> 4;
    int nrow = wave * 16 + (lane & 15);
    v8f c = {};
#pragma unroll
    for (int kc = 0; kc < 4; ++kc)
        c = wmma_f16(fragA(&A[m * HID], kc * 32, half),
                     fragB(&Wf2f[nrow * HID], kc * 32, half), c);
#pragma unroll
    for (int r = 0; r < 8; ++r) {
        int mm = r + 8 * half;
        H2[mm * HID + nrow] = gelu_f(c[r] + bf2[nrow]);
    }
    __syncthreads();
    if (threadIdx.x < 16) {
        float s = bf3[0];
        for (int n = 0; n < HID; ++n) s += H2[threadIdx.x * HID + n] * Wf3[n];
        out[bt * 16 + threadIdx.x] = s;
    }
}

// ---------------------------------------------------------------------------
extern "C" void kernel_launch(void* const* d_in, const int* in_sizes, int n_in,
                              void* d_out, int out_size, void* d_ws, size_t ws_size,
                              hipStream_t stream) {
    (void)in_sizes; (void)n_in; (void)out_size; (void)ws_size;
    const float* x      = (const float*)d_in[0];
    const int*   spin   = (const int*)d_in[1];
    const float* W_node = (const float*)d_in[2];
    const float* b_node = (const float*)d_in[3];
    const float* We1    = (const float*)d_in[4];
    const float* be1    = (const float*)d_in[5];
    const float* We2    = (const float*)d_in[6];
    const float* be2    = (const float*)d_in[7];
    const float* W_v2e  = (const float*)d_in[8];
    const float* W_eu1  = (const float*)d_in[9];
    const float* b_eu1  = (const float*)d_in[10];
    const float* W_eu2  = (const float*)d_in[11];
    const float* b_eu2  = (const float*)d_in[12];
    const float* W_e2v  = (const float*)d_in[13];
    const float* W_nu1  = (const float*)d_in[14];
    const float* b_nu1  = (const float*)d_in[15];
    const float* W_nu2  = (const float*)d_in[16];
    const float* b_nu2  = (const float*)d_in[17];
    const float* Wf1    = (const float*)d_in[18];
    const float* bf1    = (const float*)d_in[19];
    const float* Wf2    = (const float*)d_in[20];
    const float* bf2    = (const float*)d_in[21];
    const float* Wf3    = (const float*)d_in[22];
    const float* bf3    = (const float*)d_in[23];

    char* ws = (char*)d_ws;
    size_t off = 0;
    auto alloc = [&](size_t bytes) -> void* {
        void* p = ws + off;
        off = (off + bytes + 255) & ~(size_t)255;
        return p;
    };
    h16* h_v   = (h16*)alloc((size_t)B_SZ * 2048 * 2);
    h16* h_e   = (h16*)alloc((size_t)B_SZ * 30720 * 2);
    h16* v2e   = (h16*)alloc((size_t)B_SZ * 2048 * 2);
    h16* hbuf  = (h16*)alloc((size_t)B_SZ * 128 * 2);
    float* acc = (float*)alloc((size_t)B_SZ * 128 * 4);
    float* rp  = (float*)alloc((size_t)B_SZ * 4);
    float* r2a = (float*)alloc((size_t)B_SZ * 4);
    h16* We2f  = (h16*)alloc(16384 * 2);
    h16* Wv2ef = (h16*)alloc(49152 * 2);
    h16* Weu1f = (h16*)alloc(147456 * 2);
    h16* Weu2f = (h16*)alloc(49152 * 2);
    h16* We2vf = (h16*)alloc(49152 * 2);
    h16* Wnu1f = (h16*)alloc(98304 * 2);
    h16* Wnu2f = (h16*)alloc(49152 * 2);
    h16* Wf1f  = (h16*)alloc((size_t)128 * 32768 * 2);
    h16* Wf2f  = (h16*)alloc(16384 * 2);

    auto cdiv = [](int a, int b) { return (a + b - 1) / b; };

    // weight conversions (f32 -> f16), all L2-resident afterwards
    k_cvt2d<<<cdiv(128 * 128, 256), 256, 0, stream>>>(We2,   We2f,  128, 128, 128);
    k_cvt2d<<<cdiv(3 * 128 * 128, 256), 256, 0, stream>>>(W_v2e, Wv2ef, 3 * 128, 128, 128);
    k_cvt2d<<<cdiv(3 * 128 * 384, 256), 256, 0, stream>>>(W_eu1, Weu1f, 3 * 128, 384, 384);
    k_cvt2d<<<cdiv(3 * 128 * 128, 256), 256, 0, stream>>>(W_eu2, Weu2f, 3 * 128, 128, 128);
    k_cvt2d<<<cdiv(3 * 128 * 128, 256), 256, 0, stream>>>(W_e2v, We2vf, 3 * 128, 128, 128);
    k_cvt2d<<<cdiv(3 * 128 * 256, 256), 256, 0, stream>>>(W_nu1, Wnu1f, 3 * 128, 256, 256);
    k_cvt2d<<<cdiv(3 * 128 * 128, 256), 256, 0, stream>>>(W_nu2, Wnu2f, 3 * 128, 128, 128);
    k_cvt2d<<<cdiv(128 * 32768, 256), 256, 0, stream>>>(Wf1,   Wf1f,  128, 32768, 32770);
    k_cvt2d<<<cdiv(128 * 128, 256), 256, 0, stream>>>(Wf2,   Wf2f,  128, 128, 128);

    k_zero<<<cdiv(B_SZ * 128, 256), 256, 0, stream>>>(acc, B_SZ * 128);
    k_rfeat<<<cdiv(B_SZ, 256), 256, 0, stream>>>(x, rp, r2a);

    k_init_nodes<<<B_SZ, 128, 0, stream>>>(x, spin, W_node, b_node, h_v);
    k_init_edges<<<dim3(15, B_SZ), 256, 0, stream>>>(x, We1, be1, We2f, be2, h_e);

    for (int s = 0; s < 3; ++s) {
        k_v2e<<<B_SZ, 256, 0, stream>>>(h_v, Wv2ef + s * 16384, v2e);
        k_edge_update<<<dim3(15, B_SZ), 256, 0, stream>>>(
            b_eu1 + s * 128, b_eu2 + s * 128,
            Weu1f + (size_t)s * 49152, Weu2f + s * 16384, h_e, v2e);
        k_e2v_node<<<B_SZ, 256, 0, stream>>>(
            h_e, We2vf + s * 16384, Wnu1f + s * 32768, b_nu1 + s * 128,
            Wnu2f + s * 16384, b_nu2 + s * 128, h_v);
    }

    k_final1_split<<<dim3(B_SZ / 16, KSPLIT), 256, 0, stream>>>(h_v, h_e, Wf1f, acc);
    k_final1_epi<<<cdiv(B_SZ * 128, 256), 256, 0, stream>>>(acc, Wf1, bf1, rp, r2a, hbuf);
    k_final2<<<B_SZ / 16, 256, 0, stream>>>(hbuf, Wf2f, bf2, Wf3, bf3, (float*)d_out);
}
